// CISTransformerDecoderV2_43336220016861
// MI455X (gfx1250) — matrix operand order repaired and checked by reading, not compiled, not run on hardware
//
#include <hip/hip_runtime.h>

// ---------------- problem constants ----------------
#define DM   256
#define NH   8
#define LVN  4
#define NP   4
#define DFF  1024
#define NLAY 6
#define KQ   300
#define BB   2
#define STOT 21760            // 128^2+64^2+32^2+16^2
#define ROWS (BB*KQ)          // 600
#define RPAD 608              // padded to multiple of 16 for WMMA tiles
#define MROW (BB*STOT)        // 43520 (divisible by 16)

typedef __attribute__((ext_vector_type(16))) __bf16 v16bf;
typedef __attribute__((ext_vector_type(8)))  float  v8f;
typedef __attribute__((ext_vector_type(4)))  int    v4i;

// CDNA5 async global->LDS path (guarded; falls back to plain LDS copy)
#if defined(__gfx1250__) && __has_builtin(__builtin_amdgcn_global_load_async_to_lds_b128)
#define CIS_ASYNC 1
#else
#define CIS_ASYNC 0
#endif

#if CIS_ASYNC
__device__ __forceinline__ void cis_async_b128(const void* g, void* l) {
  auto gp = (__attribute__((address_space(1))) v4i*)(unsigned long long)(uintptr_t)g;
  auto lp = (__attribute__((address_space(3))) v4i*)(unsigned)(uintptr_t)l;
  __builtin_amdgcn_global_load_async_to_lds_b128(gp, lp, 0, 0);
}
__device__ __forceinline__ void cis_wait_async() {
#if __has_builtin(__builtin_amdgcn_s_wait_asynccnt)
  __builtin_amdgcn_s_wait_asynccnt(0);
#else
  asm volatile("s_wait_asynccnt 0x0" ::: "memory");
#endif
}
#endif

// fp32 -> bf16 (round-to-nearest-even)
static __device__ __forceinline__ unsigned short f2bf(float f) {
  unsigned int u = __float_as_uint(f);
  u += 0x7FFFu + ((u >> 16) & 1u);
  return (unsigned short)(u >> 16);
}

// token index -> level decode (levels: 128,64,32,16)
static __device__ __forceinline__ void decode_s(int s, int& lvl, int& dim, int& sl) {
  if (s < 16384)      { lvl = 0; dim = 128; sl = s; }
  else if (s < 20480) { lvl = 1; dim = 64;  sl = s - 16384; }
  else if (s < 21504) { lvl = 2; dim = 32;  sl = s - 20480; }
  else                { lvl = 3; dim = 16;  sl = s - 21504; }
}

// ---------------- build value_in (bf16) = flat(mem) + flat(posm) + level_embed ----------------
__global__ void cis_build_value(const float* __restrict__ m0, const float* __restrict__ m1,
                                const float* __restrict__ m2, const float* __restrict__ m3,
                                const float* __restrict__ q0, const float* __restrict__ q1,
                                const float* __restrict__ q2, const float* __restrict__ q3,
                                const float* __restrict__ emb,
                                unsigned short* __restrict__ vin) {
  size_t i = (size_t)blockIdx.x * 256 + threadIdx.x;
  if (i >= (size_t)MROW * DM) return;
  int d = (int)(i & 255);
  size_t bs = i >> 8;
  int s = (int)(bs % STOT);
  int b = (int)(bs / STOT);
  int lvl, dim, sl; decode_s(s, lvl, dim, sl);
  const float* mp = (lvl == 0) ? m0 : (lvl == 1) ? m1 : (lvl == 2) ? m2 : m3;
  const float* pp = (lvl == 0) ? q0 : (lvl == 1) ? q1 : (lvl == 2) ? q2 : q3;
  size_t o = (size_t)(b * DM + d) * dim * dim + sl;   // [B,D,h,w] layout
  vin[i] = f2bf(mp[o] + pp[o] + emb[lvl * DM + d]);
}

// ---------------- reference points per token ----------------
__global__ void cis_ref(float* __restrict__ ref) {
  int s = blockIdx.x * 256 + threadIdx.x;
  if (s >= STOT) return;
  int lvl, dim, sl; decode_s(s, lvl, dim, sl);
  int y = sl / dim, x = sl % dim;
  ref[s * 2 + 0] = (x + 0.5f) / (float)dim;
  ref[s * 2 + 1] = (y + 0.5f) / (float)dim;
}

// ---------------- scores = feat @ qs_w + qs_b (wave per token) ----------------
__global__ void cis_scores(const float* __restrict__ s0, const float* __restrict__ s1,
                           const float* __restrict__ s2, const float* __restrict__ s3,
                           const float* __restrict__ qsw, const float* __restrict__ qsb,
                           float* __restrict__ scores) {
  int idx = blockIdx.x * 8 + (threadIdx.x >> 5);
  if (idx >= MROW) return;
  int lane = threadIdx.x & 31;
  int b = idx / STOT, s = idx % STOT;
  int lvl, dim, sl; decode_s(s, lvl, dim, sl);
  const float* sp = (lvl == 0) ? s0 : (lvl == 1) ? s1 : (lvl == 2) ? s2 : s3;
  float acc = 0.f;
  for (int d = lane; d < DM; d += 32) {
    acc += sp[(size_t)(b * DM + d) * dim * dim + sl] * qsw[d];
  }
  for (int o = 16; o > 0; o >>= 1) acc += __shfl_down(acc, o);
  if (lane == 0) scores[idx] = acc + qsb[0];
}

// ---------------- top-K=300 via iterative argmax (one block per batch) ----------------
__global__ void cis_topk(float* __restrict__ sc, int* __restrict__ out) {
  __shared__ float sv[1024];
  __shared__ int   si[1024];
  int b = blockIdx.x, t = threadIdx.x;
  float* s = sc + (size_t)b * STOT;
  for (int k = 0; k < KQ; ++k) {
    float best = -3.4e38f; int bi = 0x7fffffff;
    for (int i = t; i < STOT; i += 1024) {
      float v = s[i];
      if (v > best || (v == best && i < bi)) { best = v; bi = i; }
    }
    sv[t] = best; si[t] = bi;
    __syncthreads();
    for (int o = 512; o > 0; o >>= 1) {
      if (t < o) {
        if (sv[t + o] > sv[t] || (sv[t + o] == sv[t] && si[t + o] < si[t])) {
          sv[t] = sv[t + o]; si[t] = si[t + o];
        }
      }
      __syncthreads();
    }
    if (t == 0) { out[b * KQ + k] = si[0]; s[si[0]] = -3.4e38f; }
    __syncthreads();
  }
}

// ---------------- gather x / q_pos / q_ref at top-k tokens (pad rows zeroed) ----------------
__global__ void cis_gather(const float* __restrict__ s0, const float* __restrict__ s1,
                           const float* __restrict__ s2, const float* __restrict__ s3,
                           const float* __restrict__ p0, const float* __restrict__ p1,
                           const float* __restrict__ p2, const float* __restrict__ p3,
                           const float* __restrict__ emb, const int* __restrict__ topk,
                           const float* __restrict__ ref,
                           float* __restrict__ x, float* __restrict__ qpos,
                           float* __restrict__ qref) {
  int r = blockIdx.x, t = threadIdx.x;
  if (r >= ROWS) {
    x[(size_t)r * DM + t] = 0.f; qpos[(size_t)r * DM + t] = 0.f;
    if (t < 2) qref[r * 2 + t] = 0.f;
    return;
  }
  int b = r / KQ;
  int s = topk[r];
  int lvl, dim, sl; decode_s(s, lvl, dim, sl);
  const float* sp = (lvl == 0) ? s0 : (lvl == 1) ? s1 : (lvl == 2) ? s2 : s3;
  const float* pp = (lvl == 0) ? p0 : (lvl == 1) ? p1 : (lvl == 2) ? p2 : p3;
  size_t o = (size_t)(b * DM + t) * dim * dim + sl;
  x[(size_t)r * DM + t]    = sp[o];
  qpos[(size_t)r * DM + t] = pp[o] + emb[lvl * DM + t];
  if (t < 2) qref[r * 2 + t] = ref[s * 2 + t];
}

// ---------------- pack fp32 [K][N] weights -> WMMA B-fragment order ----------------
// dword index i = ((kb*N + col)*2 + half)*8 + r  holds bf16 pair (k, k+1),
// k = kb*32 + half*16 + 2r.  Per lane per k-step the 8 dwords are contiguous (2 x b128).
__global__ void cis_pack(const float* __restrict__ W, unsigned int* __restrict__ Bp, int K, int N) {
  int i = blockIdx.x * 256 + threadIdx.x;
  int tot = (K >> 1) * N;
  if (i >= tot) return;
  int r  = i & 7;
  int t  = i >> 3;
  int hf = t & 1;  t >>= 1;
  int c  = t % N;
  int kb = t / N;
  int k  = kb * 32 + hf * 16 + 2 * r;
  unsigned int lo = f2bf(W[(size_t)k * N + c]);
  unsigned int hi = f2bf(W[(size_t)(k + 1) * N + c]);
  Bp[i] = lo | (hi << 16);
}

// ---------------- fp32 -> bf16 convert with zero-padding beyond `valid` ----------------
__global__ void cis_cvt(const float* __restrict__ src, unsigned short* __restrict__ dst,
                        int total, int valid) {
  int i = blockIdx.x * 256 + threadIdx.x;
  if (i >= total) return;
  dst[i] = (i < valid) ? f2bf(src[i]) : (unsigned short)0;
}

// ---------------- elementwise add (qp = x + q_pos) ----------------
__global__ void cis_add(const float* __restrict__ a, const float* __restrict__ b,
                        float* __restrict__ c, int n) {
  int i = blockIdx.x * 256 + threadIdx.x;
  if (i < n) c[i] = a[i] + b[i];
}

// ---------------- bf16 WMMA GEMM: C[M,N]f32 = A[M,K]bf16 @ B + bias, opt relu ----------------
// 128 threads = 4 waves; wave tile 16(M) x 32(N) (2 WMMA accumulators, shared A frag);
// block tile 16 x 128. A strip staged in LDS (async global->LDS when available).
template <int N, int K>
__global__ void cis_gemm_bf16(const unsigned short* __restrict__ A,
                              const unsigned int* __restrict__ Bp,   // fragment-packed
                              const float* __restrict__ bias,
                              float* __restrict__ C,
                              int relu) {
  __shared__ unsigned short sA[16 * K];
  const int tid  = threadIdx.x;
  const int wave = tid >> 5;
  const int lane = tid & 31;
  const int m0 = blockIdx.y * 16;
  const int n0 = blockIdx.x * 128 + wave * 32;

  // ---- stage 16 x K strip of A into LDS ----
  constexpr int KV = K >> 3;                 // uint4 per row
#if CIS_ASYNC
  for (int i = tid; i < 16 * KV; i += 128) {
    int row = i / KV, c = i % KV;
    cis_async_b128(reinterpret_cast<const uint4*>(A + (size_t)(m0 + row) * K) + c,
                   reinterpret_cast<uint4*>(sA) + i);
  }
  cis_wait_async();
  __syncthreads();
#else
  for (int i = tid; i < 16 * KV; i += 128) {
    int row = i / KV, c = i % KV;
    reinterpret_cast<uint4*>(sA)[i] =
        reinterpret_cast<const uint4*>(A + (size_t)(m0 + row) * K)[c];
  }
  __syncthreads();
#endif

  v8f acc0 = {0.f, 0.f, 0.f, 0.f, 0.f, 0.f, 0.f, 0.f};
  v8f acc1 = {0.f, 0.f, 0.f, 0.f, 0.f, 0.f, 0.f, 0.f};
  const int half8 = (lane >= 16) ? 8 : 0;           // A K-half in elems
  const int hf    = (lane >= 16) ? 1 : 0;           // B half bit
  const int rowA  = lane & 15;
  const int col0  = n0 + (lane & 15);

  // B fragment base (uint4 units); per-kblock stride is the compile-time constant N*4
  const uint4* __restrict__ Bv = reinterpret_cast<const uint4*>(Bp);
  size_t bidx = ((size_t)col0 * 2 + hf) << 1;

#pragma unroll 4
  for (int kk = 0; kk < K; kk += 32, bidx += (size_t)N * 4) {
    union { uint4 q[2]; v16bf v; } afr, bf0, bf1;
    afr.q[0] = *reinterpret_cast<const uint4*>(&sA[(size_t)rowA * K + kk + 2 * half8]);
    afr.q[1] = *reinterpret_cast<const uint4*>(&sA[(size_t)rowA * K + kk + 2 * half8 + 8]);
    bf0.q[0] = Bv[bidx];      bf0.q[1] = Bv[bidx + 1];
    bf1.q[0] = Bv[bidx + 64]; bf1.q[1] = Bv[bidx + 65];   // +16 cols = +1024B immediate
    if (kk + 32 < K) __builtin_prefetch(Bv + bidx + (size_t)N * 4, 0, 0);
    acc0 = __builtin_amdgcn_wmma_f32_16x16x32_bf16(false, afr.v, false, bf0.v,
                                                   (short)0, acc0, false, false);
    acc1 = __builtin_amdgcn_wmma_f32_16x16x32_bf16(false, afr.v, false, bf1.v,
                                                   (short)0, acc1, false, false);
  }

  // ---- epilogue: lanes 0-15 rows m0+0..7, lanes 16-31 rows m0+8..15 ----
  const int mb = m0 + ((lane < 16) ? 0 : 8);
  const float bv0 = bias ? bias[col0] : 0.f;
  const float bv1 = bias ? bias[col0 + 16] : 0.f;
#pragma unroll
  for (int r = 0; r < 8; ++r) {
    float v0 = acc0[r] + bv0;
    float v1 = acc1[r] + bv1;
    if (relu) { v0 = fmaxf(v0, 0.f); v1 = fmaxf(v1, 0.f); }
    C[(size_t)(mb + r) * N + col0]      = v0;
    C[(size_t)(mb + r) * N + col0 + 16] = v1;
  }
}

// ---------------- softmax over LV*P=16 sampling weights per (row, head) ----------------
__global__ void cis_softmax(float* __restrict__ aw) {
  int i = blockIdx.x * 256 + threadIdx.x;
  if (i >= ROWS * NH) return;
  float* p = aw + (size_t)i * 16;   // [r][h][16] contiguous
  float mx = p[0];
#pragma unroll
  for (int j = 1; j < 16; ++j) mx = fmaxf(mx, p[j]);
  float sum = 0.f;
  float e[16];
#pragma unroll
  for (int j = 0; j < 16; ++j) { e[j] = __expf(p[j] - mx); sum += e[j]; }
  float inv = 1.f / sum;
#pragma unroll
  for (int j = 0; j < 16; ++j) p[j] = e[j] * inv;
}

// ---------------- multi-scale deformable sampling: wave = (b,k,h), lane = channel ----------------
__global__ void cis_sample(const float* __restrict__ value,   // [B*S][256] = (B,S,H,32)
                           const float* __restrict__ off,     // [RPAD][256]
                           const float* __restrict__ aw,      // [RPAD][128]
                           const float* __restrict__ qref,    // [RPAD][2]
                           float* __restrict__ attn) {        // [RPAD][256]
  int wid = blockIdx.x * 8 + (threadIdx.x >> 5);
  if (wid >= ROWS * NH) return;
  int lane = threadIdx.x & 31;
  int r = wid / NH, h = wid % NH;
  int b = r / KQ;
  float rx = qref[r * 2 + 0], ry = qref[r * 2 + 1];
  const int dims[4]   = {128, 64, 32, 16};
  const int starts[4] = {0, 16384, 20480, 21504};
  float acc = 0.f;
#pragma unroll
  for (int lvl = 0; lvl < LVN; ++lvl) {
    int w = dims[lvl], st = starts[lvl];
    float fw = (float)w;
#pragma unroll
    for (int p = 0; p < NP; ++p) {
      int oi = ((r * NH + h) * LVN + lvl) * NP + p;
      float a  = aw[oi];
      float X  = (rx + off[oi * 2 + 0] / fw) * fw - 0.5f;
      float Y  = (ry + off[oi * 2 + 1] / fw) * fw - 0.5f;
      float x0 = floorf(X), y0 = floorf(Y);
#pragma unroll
      for (int dx = 0; dx < 2; ++dx) {
#pragma unroll
        for (int dy = 0; dy < 2; ++dy) {
          float xi = x0 + dx, yi = y0 + dy;
          float wgt = (1.f - fabsf(X - xi)) * (1.f - fabsf(Y - yi));
          bool valid = (xi >= 0.f) & (xi <= fw - 1.f) & (yi >= 0.f) & (yi <= fw - 1.f);
          if (valid) {
            int xc = min(max((int)xi, 0), w - 1);
            int yc = min(max((int)yi, 0), w - 1);
            size_t row = (size_t)b * STOT + st + yc * w + xc;
            acc += value[row * DM + h * 32 + lane] * wgt * a;
          }
        }
      }
    }
  }
  attn[(size_t)r * DM + h * 32 + lane] = acc;
}

// ---------------- LayerNorm: x = LN(x + a) * s + b (block per row) ----------------
__global__ void cis_ln(float* __restrict__ x, const float* __restrict__ a,
                       const float* __restrict__ sc, const float* __restrict__ bi) {
  __shared__ float red[256];
  int r = blockIdx.x, t = threadIdx.x;
  float v = x[(size_t)r * DM + t] + a[(size_t)r * DM + t];
  red[t] = v; __syncthreads();
  for (int o = 128; o > 0; o >>= 1) { if (t < o) red[t] += red[t + o]; __syncthreads(); }
  float m = red[0] * (1.f / DM); __syncthreads();
  float d = v - m;
  red[t] = d * d; __syncthreads();
  for (int o = 128; o > 0; o >>= 1) { if (t < o) red[t] += red[t + o]; __syncthreads(); }
  float var = red[0] * (1.f / DM);
  x[(size_t)r * DM + t] = d * rsqrtf(var + 1e-5f) * sc[t] + bi[t];
}

// ---------------- final output: x rows + lsi constants ----------------
__global__ void cis_out(const float* __restrict__ x, float* __restrict__ out) {
  int i = blockIdx.x * 256 + threadIdx.x;
  if (i < ROWS * DM) out[i] = x[i];
  else if (i < ROWS * DM + 4) {
    const int lsi[4] = {0, 16384, 20480, 21504};
    out[i] = (float)lsi[i - ROWS * DM];
  }
}

// =====================================================================================
extern "C" void kernel_launch(void* const* d_in, const int* in_sizes, int n_in,
                              void* d_out, int out_size, void* d_ws, size_t ws_size,
                              hipStream_t stream) {
  (void)in_sizes; (void)n_in; (void)out_size; (void)ws_size;
  const float* SRC[4]  = {(const float*)d_in[0],  (const float*)d_in[4],  (const float*)d_in[8],  (const float*)d_in[12]};
  const float* POS[4]  = {(const float*)d_in[1],  (const float*)d_in[5],  (const float*)d_in[9],  (const float*)d_in[13]};
  const float* MEM[4]  = {(const float*)d_in[2],  (const float*)d_in[6],  (const float*)d_in[10], (const float*)d_in[14]};
  const float* POSM[4] = {(const float*)d_in[3],  (const float*)d_in[7],  (const float*)d_in[11], (const float*)d_in[15]};
  const float* emb   = (const float*)d_in[16];
  const float* qs_w  = (const float*)d_in[17];
  const float* qs_b  = (const float*)d_in[18];
  const float* off_w = (const float*)d_in[19];
  const float* off_b = (const float*)d_in[20];
  const float* aw_w  = (const float*)d_in[21];
  const float* aw_b  = (const float*)d_in[22];
  const float* val_w = (const float*)d_in[23];
  const float* val_b = (const float*)d_in[24];
  const float* op_w  = (const float*)d_in[25];
  const float* op_b  = (const float*)d_in[26];
  const float* n1_s  = (const float*)d_in[27];
  const float* n1_b  = (const float*)d_in[28];
  const float* f1_w  = (const float*)d_in[29];
  const float* f1_b  = (const float*)d_in[30];
  const float* f2_w  = (const float*)d_in[31];
  const float* f2_b  = (const float*)d_in[32];
  const float* n2_s  = (const float*)d_in[33];
  const float* n2_b  = (const float*)d_in[34];

  // -------- workspace layout (256B aligned slices) --------
  char* base = (char*)d_ws;
  size_t cur = 0;
  auto alloc = [&](size_t bytes) -> char* {
    char* p = base + cur;
    cur += (bytes + 255) & ~(size_t)255;
    return p;
  };
  unsigned short* vin_bf = (unsigned short*)alloc((size_t)MROW * DM * 2);
  float* value  = (float*)alloc((size_t)MROW * DM * 4);
  float* ref    = (float*)alloc((size_t)STOT * 2 * 4);
  float* scores = (float*)alloc((size_t)MROW * 4);
  int*   topk   = (int*)  alloc((size_t)ROWS * 4);
  float* x      = (float*)alloc((size_t)RPAD * DM * 4);
  float* qpos   = (float*)alloc((size_t)RPAD * DM * 4);
  float* qp     = (float*)alloc((size_t)RPAD * DM * 4);
  float* offb   = (float*)alloc((size_t)RPAD * DM * 4);
  float* awb    = (float*)alloc((size_t)RPAD * 128 * 4);
  float* attn   = (float*)alloc((size_t)RPAD * DM * 4);
  float* aout   = (float*)alloc((size_t)RPAD * DM * 4);
  float* h1     = (float*)alloc((size_t)RPAD * DFF * 4);
  float* f2o    = (float*)alloc((size_t)RPAD * DM * 4);
  float* qref   = (float*)alloc((size_t)RPAD * 2 * 4);
  unsigned short* qp_bf   = (unsigned short*)alloc((size_t)RPAD * DM * 2);
  unsigned short* x_bf    = (unsigned short*)alloc((size_t)RPAD * DM * 2);
  unsigned short* attn_bf = (unsigned short*)alloc((size_t)RPAD * DM * 2);
  unsigned short* h1_bf   = (unsigned short*)alloc((size_t)RPAD * DFF * 2);
  unsigned int* offp = (unsigned int*)alloc((size_t)NLAY * 128 * 256 * 4);
  unsigned int* awp  = (unsigned int*)alloc((size_t)NLAY * 128 * 128 * 4);
  unsigned int* valp = (unsigned int*)alloc((size_t)NLAY * 128 * 256 * 4);
  unsigned int* opp  = (unsigned int*)alloc((size_t)NLAY * 128 * 256 * 4);
  unsigned int* f1p  = (unsigned int*)alloc((size_t)NLAY * 128 * 1024 * 4);
  unsigned int* f2p  = (unsigned int*)alloc((size_t)NLAY * 512 * 256 * 4);

  // -------- stage 0: value_in, refs, scores, top-k, gather --------
  cis_build_value<<<(MROW * DM + 255) / 256, 256, 0, stream>>>(
      MEM[0], MEM[1], MEM[2], MEM[3], POSM[0], POSM[1], POSM[2], POSM[3], emb, vin_bf);
  cis_ref<<<(STOT + 255) / 256, 256, 0, stream>>>(ref);
  cis_scores<<<MROW / 8, 256, 0, stream>>>(SRC[0], SRC[1], SRC[2], SRC[3], qs_w, qs_b, scores);
  cis_topk<<<BB, 1024, 0, stream>>>(scores, topk);
  cis_gather<<<RPAD, 256, 0, stream>>>(SRC[0], SRC[1], SRC[2], SRC[3],
                                       POS[0], POS[1], POS[2], POS[3],
                                       emb, topk, ref, x, qpos, qref);

  // -------- stage 1: pack all layer weights into WMMA fragment layout --------
  for (int l = 0; l < NLAY; ++l) {
    cis_pack<<<(128 * 256 + 255) / 256, 256, 0, stream>>>(off_w + (size_t)l * 256 * 256, offp + (size_t)l * 128 * 256, 256, 256);
    cis_pack<<<(128 * 128 + 255) / 256, 256, 0, stream>>>(aw_w  + (size_t)l * 256 * 128, awp  + (size_t)l * 128 * 128, 256, 128);
    cis_pack<<<(128 * 256 + 255) / 256, 256, 0, stream>>>(val_w + (size_t)l * 256 * 256, valp + (size_t)l * 128 * 256, 256, 256);
    cis_pack<<<(128 * 256 + 255) / 256, 256, 0, stream>>>(op_w  + (size_t)l * 256 * 256, opp  + (size_t)l * 128 * 256, 256, 256);
    cis_pack<<<(128 * 1024 + 255) / 256, 256, 0, stream>>>(f1_w + (size_t)l * 256 * 1024, f1p + (size_t)l * 128 * 1024, 256, 1024);
    cis_pack<<<(512 * 256 + 255) / 256, 256, 0, stream>>>(f2_w + (size_t)l * 1024 * 256, f2p + (size_t)l * 512 * 256, 1024, 256);
  }

  // -------- stage 2: decoder layers --------
  for (int l = 0; l < NLAY; ++l) {
    cis_add<<<(ROWS * DM + 255) / 256, 256, 0, stream>>>(x, qpos, qp, ROWS * DM);
    cis_cvt<<<(RPAD * DM + 255) / 256, 256, 0, stream>>>(qp, qp_bf, RPAD * DM, ROWS * DM);

    // sampling offsets & attention logits (WMMA)
    cis_gemm_bf16<256, 256><<<dim3(2, RPAD / 16), 128, 0, stream>>>(
        qp_bf, offp + (size_t)l * 128 * 256, off_b + (size_t)l * 256, offb, 0);
    cis_gemm_bf16<128, 256><<<dim3(1, RPAD / 16), 128, 0, stream>>>(
        qp_bf, awp + (size_t)l * 128 * 128, aw_b + (size_t)l * 128, awb, 0);
    cis_softmax<<<(ROWS * NH + 255) / 256, 256, 0, stream>>>(awb);

    // value projection over full sequence -- the dominant GEMM (WMMA)
    cis_gemm_bf16<256, 256><<<dim3(2, MROW / 16), 128, 0, stream>>>(
        vin_bf, valp + (size_t)l * 128 * 256, val_b + (size_t)l * 256, value, 0);

    // bilinear deformable sampling
    cis_sample<<<(ROWS * NH) / 8, 256, 0, stream>>>(value, offb, awb, qref, attn);
    cis_cvt<<<(RPAD * DM + 255) / 256, 256, 0, stream>>>(attn, attn_bf, RPAD * DM, ROWS * DM);

    // output projection + residual LN
    cis_gemm_bf16<256, 256><<<dim3(2, RPAD / 16), 128, 0, stream>>>(
        attn_bf, opp + (size_t)l * 128 * 256, op_b + (size_t)l * 256, aout, 0);
    cis_ln<<<ROWS, 256, 0, stream>>>(x, aout, n1_s + (size_t)l * 256, n1_b + (size_t)l * 256);
    cis_cvt<<<(RPAD * DM + 255) / 256, 256, 0, stream>>>(x, x_bf, RPAD * DM, ROWS * DM);

    // FFN (relu fused) + residual LN
    cis_gemm_bf16<1024, 256><<<dim3(8, RPAD / 16), 128, 0, stream>>>(
        x_bf, f1p + (size_t)l * 128 * 1024, f1_b + (size_t)l * 1024, h1, 1);
    cis_cvt<<<(RPAD * DFF + 255) / 256, 256, 0, stream>>>(h1, h1_bf, RPAD * DFF, ROWS * DFF);
    cis_gemm_bf16<256, 1024><<<dim3(2, RPAD / 16), 128, 0, stream>>>(
        h1_bf, f2p + (size_t)l * 512 * 256, f2_b + (size_t)l * 256, f2o, 0);
    cis_ln<<<ROWS, 256, 0, stream>>>(x, f2o, n2_s + (size_t)l * 256, n2_b + (size_t)l * 256);
  }

  // -------- stage 3: emit (x, lsi) --------
  cis_out<<<(ROWS * DM + 4 + 255) / 256, 256, 0, stream>>>(x, (float*)d_out);
}
